// Q_Attention_63084479643972
// MI455X (gfx1250) — 8-bit matrix operand order repaired and checked by reading, compiled, not run on hardware
//
#include <hip/hip_runtime.h>
#include <hip/hip_bf16.h>
#include <cstdint>
#include <cstddef>

typedef int v8i __attribute__((ext_vector_type(8)));

#define HEADS 12
#define HD 64
#define DIMC 768
#define NSEQ 1024
#define BATCH 8
#define BH (BATCH*HEADS)
#define MROWS (BATCH*NSEQ)

__device__ inline v8i load_frag_a(const int8_t* base, int ld, int l16, int half) {
  v8i f;
#pragma unroll
  for (int v = 0; v < 8; ++v) {
    int ko = ((v >> 2) << 5) + ((v & 3) << 2) + (half << 4);
    f[v] = *(const int*)(base + (size_t)l16 * ld + ko);
  }
  return f;
}

__device__ inline v8i load_frag_b(const int8_t* base, int ld, int l16, int half) {
  v8i f;
#pragma unroll
  for (int v = 0; v < 8; ++v) {
    int ko = ((v >> 2) << 5) + ((v & 3) << 2) + (half << 4);
    f[v] = *(const int*)(base + (size_t)l16 * ld + ko);
  }
  return f;
}

__device__ inline int clampi(int x, int lo, int hi) {
  return x < lo ? lo : (x > hi ? hi : x);
}

__global__ void scalars_kernel(const float* qkv_aa, const float* qkv_zp,
                               const float* q_a, const float* q_z,
                               const float* k_a, const float* k_z,
                               const float* v_a, const float* v_z,
                               const float* at_a, const float* at_z,
                               const float* p_aa, const float* p_zp,
                               float* params) {
  int lane = threadIdx.x;
  auto meanN = [&](const float* p, int n, bool rnd) -> float {
    float s = 0.f;
    for (int i = lane; i < n; i += 32) s += rnd ? rintf(p[i]) : p[i];
    for (int m = 16; m >= 1; m >>= 1) s += __shfl_xor(s, m, 32);
    return s / (float)n;
  };
  float r0  = meanN(qkv_aa, DIMC, false);
  float r1  = meanN(qkv_zp, DIMC, true);
  float r2  = meanN(q_a, HEADS, false);
  float r3  = meanN(q_z, HEADS, true);
  float r4  = meanN(k_a, HEADS, false);
  float r5  = meanN(k_z, HEADS, true);
  float r6  = meanN(v_a, HEADS, false);
  float r7  = meanN(v_z, HEADS, true);
  float r8  = meanN(at_a, HEADS, false);
  float r9  = meanN(at_z, HEADS, true);
  float r10 = meanN(p_aa, DIMC, false);
  float r11 = meanN(p_zp, DIMC, true);
  if (lane == 0) {
    params[0] = r0;  params[1] = r1;  params[2] = r2;  params[3] = r3;
    params[4] = r4;  params[5] = r5;  params[6] = r6;  params[7] = r7;
    params[8] = r8;  params[9] = r9;  params[10] = r10; params[11] = r11;
  }
}

__global__ __launch_bounds__(256) void quant_w(const float* __restrict__ qkvw,
                                               const float* __restrict__ qkv_aw,
                                               const float* __restrict__ projw,
                                               const float* __restrict__ proj_aw,
                                               int8_t* __restrict__ wq_qkv,
                                               float* __restrict__ cs_qkv,
                                               int8_t* __restrict__ wq_proj,
                                               float* __restrict__ cs_proj) {
  int lane = threadIdx.x & 31, w = threadIdx.x >> 5;
  int row = blockIdx.x * 8 + w;
  const float* src; float a; int8_t* dst; float* cs;
  if (row < 3 * DIMC) {
    src = qkvw + (size_t)row * DIMC; a = qkv_aw[row];
    dst = wq_qkv + (size_t)row * DIMC; cs = cs_qkv + row;
  } else {
    int r = row - 3 * DIMC;
    src = projw + (size_t)r * DIMC; a = proj_aw[r];
    dst = wq_proj + (size_t)r * DIMC; cs = cs_proj + r;
  }
  float inv = 1.f / a;
  int s = 0;
#pragma unroll 4
  for (int j = 0; j < DIMC / 32; ++j) {
    int idx = lane + j * 32;
    int t = clampi((int)rintf(src[idx] * inv), -8, 7);
    dst[idx] = (int8_t)t;
    s += t;
  }
  for (int m = 16; m >= 1; m >>= 1) s += __shfl_xor(s, m, 32);
  if (lane == 0) *cs = (float)s;
}

__global__ void quant_x0(const float* __restrict__ x0,
                         const float* __restrict__ params, int pidx,
                         int8_t* __restrict__ xq, int n4) {
  int i = blockIdx.x * blockDim.x + threadIdx.x;
  if (i >= n4) return;
  float inv = 1.f / params[pidx], z = params[pidx + 1];
  float4 v = ((const float4*)x0)[i];
  char4 o;
  o.x = (signed char)clampi((int)rintf(v.x * inv + z), -8, 7);
  o.y = (signed char)clampi((int)rintf(v.y * inv + z), -8, 7);
  o.z = (signed char)clampi((int)rintf(v.z * inv + z), -8, 7);
  o.w = (signed char)clampi((int)rintf(v.w * inv + z), -8, 7);
  ((char4*)xq)[i] = o;
}

__global__ __launch_bounds__(256) void gemm_q4(
    const int8_t* __restrict__ A, const int8_t* __restrict__ Bw,
    const float* __restrict__ alpha, const float* __restrict__ cs,
    const float* __restrict__ bias, float* __restrict__ C,
    int Nout, int K, const float* __restrict__ params, int pidx) {
  __shared__ __align__(16) int8_t sA[128 * 64];
  __shared__ __align__(16) int8_t sB[128 * 64];
  const int tid = threadIdx.x, lane = tid & 31, w = tid >> 5;
  const int l16 = lane & 15, half = lane >> 4;
  const int nb = Nout >> 7;
  const int bm = blockIdx.x / nb, bn = blockIdx.x % nb;
  const int wm = w & 1, wn = w >> 1;
  const float a_act = params[pidx], z_act = params[pidx + 1];
  v8i acc[4][2] = {};
  const int srow = tid >> 1, soff = (tid & 1) * 32;
  for (int k0 = 0; k0 < K; k0 += 64) {
    __syncthreads();
    *(int4*)(sA + srow * 64 + soff) =
        *(const int4*)(A + (size_t)(bm * 128 + srow) * K + k0 + soff);
    *(int4*)(sA + srow * 64 + soff + 16) =
        *(const int4*)(A + (size_t)(bm * 128 + srow) * K + k0 + soff + 16);
    *(int4*)(sB + srow * 64 + soff) =
        *(const int4*)(Bw + (size_t)(bn * 128 + srow) * K + k0 + soff);
    *(int4*)(sB + srow * 64 + soff + 16) =
        *(const int4*)(Bw + (size_t)(bn * 128 + srow) * K + k0 + soff + 16);
    __syncthreads();
    v8i af[4], bf[2];
#pragma unroll
    for (int i = 0; i < 4; ++i)
      af[i] = load_frag_a(sA + (wm * 64 + i * 16) * 64, 64, l16, half);
#pragma unroll
    for (int j = 0; j < 2; ++j)
      bf[j] = load_frag_b(sB + (wn * 32 + j * 16) * 64, 64, l16, half);
#pragma unroll
    for (int i = 0; i < 4; ++i)
#pragma unroll
      for (int j = 0; j < 2; ++j)
        acc[i][j] = __builtin_amdgcn_wmma_i32_16x16x64_iu8(
            true, af[i], true, bf[j], acc[i][j], false, false);
  }
#pragma unroll
  for (int j = 0; j < 2; ++j) {
    int oc = bn * 128 + wn * 32 + j * 16 + l16;
    float al = alpha[oc] * a_act;
    float zc = z_act * cs[oc];
    float bs = bias ? bias[oc] : 0.f;
#pragma unroll
    for (int i = 0; i < 4; ++i) {
      int rbase = bm * 128 + wm * 64 + i * 16 + half * 8;
#pragma unroll
      for (int r = 0; r < 8; ++r)
        C[(size_t)(rbase + r) * Nout + oc] = al * ((float)acc[i][j][r] - zc) + bs;
    }
  }
}

__global__ __launch_bounds__(256) void ln_quant(
    const float* __restrict__ qkvf,
    const float* __restrict__ gq, const float* __restrict__ bq,
    const float* __restrict__ gk, const float* __restrict__ bk,
    const float* __restrict__ params,
    int8_t* __restrict__ q2i, int8_t* __restrict__ k2i, int8_t* __restrict__ v2t,
    float* __restrict__ rsq, float* __restrict__ rsk) {
  const int lane = threadIdx.x & 31, w = threadIdx.x >> 5;
  const int row = blockIdx.x * 8 + w;
  const int bh = row >> 10, n = row & 1023;
  const int b = bh / HEADS, h = bh % HEADS;
  const float* base = qkvf + ((size_t)(b * NSEQ + n)) * (3 * DIMC) + h * HD;
  const int d0 = lane, d1 = lane + 32;
  auto lnq = [&](const float* src, const float* g, const float* be,
                 float a, float z, int8_t* dst, float* rs) {
    float x0 = src[d0], x1 = src[d1];
    float s = x0 + x1;
    for (int m = 16; m >= 1; m >>= 1) s += __shfl_xor(s, m, 32);
    float mean = s * (1.f / 64.f);
    float e0 = x0 - mean, e1 = x1 - mean;
    float vs = e0 * e0 + e1 * e1;
    for (int m = 16; m >= 1; m >>= 1) vs += __shfl_xor(vs, m, 32);
    float inv = rsqrtf(vs * (1.f / 64.f) + 1e-5f);
    float y0 = e0 * inv * g[d0] + be[d0];
    float y1 = e1 * inv * g[d1] + be[d1];
    float ia = 1.f / a;
    int q0 = clampi((int)rintf(y0 * ia + z), -8, 7);
    int q1 = clampi((int)rintf(y1 * ia + z), -8, 7);
    dst[d0] = (int8_t)q0; dst[d1] = (int8_t)q1;
    int rsum = q0 + q1;
    for (int m = 16; m >= 1; m >>= 1) rsum += __shfl_xor(rsum, m, 32);
    if (lane == 0) *rs = (float)rsum;
  };
  size_t rowoff = ((size_t)bh * NSEQ + n);
  lnq(base,        gq, bq, params[2], params[3], q2i + rowoff * HD, rsq + rowoff);
  lnq(base + DIMC, gk, bk, params[4], params[5], k2i + rowoff * HD, rsk + rowoff);
  {
    float iv = 1.f / params[6], zv = params[7];
    float x0 = base[2 * DIMC + d0], x1 = base[2 * DIMC + d1];
    int q0 = clampi((int)rintf(x0 * iv + zv), -8, 7);
    int q1 = clampi((int)rintf(x1 * iv + zv), -8, 7);
    int8_t* vb = v2t + (size_t)bh * HD * NSEQ;
    vb[(size_t)d0 * NSEQ + n] = (int8_t)q0;
    vb[(size_t)d1 * NSEQ + n] = (int8_t)q1;
  }
}

__global__ __launch_bounds__(256) void colsum_v(const int8_t* __restrict__ v2t,
                                                int* __restrict__ csv) {
  int lane = threadIdx.x & 31, w = threadIdx.x >> 5;
  int row = blockIdx.x * 8 + w;
  const int8_t* p = v2t + (size_t)row * NSEQ;
  int s = 0;
#pragma unroll 8
  for (int j = 0; j < NSEQ / 32; ++j) s += p[lane + j * 32];
  for (int m = 16; m >= 1; m >>= 1) s += __shfl_xor(s, m, 32);
  if (lane == 0) csv[row] = s;
}

__global__ __launch_bounds__(256) void attn_q4(
    const int8_t* __restrict__ q2i, const int8_t* __restrict__ k2i,
    const int8_t* __restrict__ v2t, const float* __restrict__ rsq,
    const float* __restrict__ rsk, const int* __restrict__ csv,
    const float* __restrict__ params, int8_t* __restrict__ x1q) {
  __shared__ __align__(16) int8_t sK[64 * 64];
  __shared__ __align__(16) int8_t sV[64 * 64];
  __shared__ __align__(16) uint8_t sP[8 * 16 * 64];
  const int tid = threadIdx.x, lane = tid & 31, w = tid >> 5;
  const int l16 = lane & 15, half = lane >> 4;
  const int qt = blockIdx.x & 7, bh = blockIdx.x >> 3;
  const int b = bh / HEADS, h = bh % HEADS;
  const float z_q = params[3], z_k = params[5];
  const float a_v = params[6], z_v = params[7];
  const float a_at = params[8], z_at = params[9];
  const float a_p = params[10], z_p = params[11];
  const float sqk = 0.125f * params[2] * params[4];
  const float zz = 64.f * z_q * z_k;
  const int qrow0 = qt * 128 + w * 16;
  const int8_t* qbase = q2i + ((size_t)bh * NSEQ + qrow0) * HD;
  const int8_t* kbase = k2i + (size_t)bh * NSEQ * HD;
  const int8_t* vbase = v2t + (size_t)bh * HD * NSEQ;
  const int srow = tid >> 2, soff = (tid & 3) * 16;

  v8i aq = load_frag_a(qbase, HD, l16, half);
  float rq[8];
#pragma unroll
  for (int r = 0; r < 8; ++r) rq[r] = rsq[(size_t)bh * NSEQ + qrow0 + half * 8 + r];
  float mrow[8], lrow[8];
#pragma unroll
  for (int r = 0; r < 8; ++r) { mrow[r] = -INFINITY; lrow[r] = 0.f; }

  for (int kt = 0; kt < NSEQ / 64; ++kt) {
    const int kb = kt * 64;
    __syncthreads();
    *(int4*)(sK + srow * 64 + soff) =
        *(const int4*)(kbase + (size_t)(kb + srow) * HD + soff);
    __syncthreads();
    v8i cf[4];
#pragma unroll
    for (int nf = 0; nf < 4; ++nf) {
      v8i bk = load_frag_b(sK + nf * 16 * 64, 64, l16, half);
      v8i z = {};
      cf[nf] = __builtin_amdgcn_wmma_i32_16x16x64_iu8(true, aq, true, bk, z,
                                                      false, false);
    }
    float rk[4];
#pragma unroll
    for (int nf = 0; nf < 4; ++nf)
      rk[nf] = rsk[(size_t)bh * NSEQ + kb + nf * 16 + l16];
#pragma unroll
    for (int r = 0; r < 8; ++r) {
      float sv[4]; float smax = -INFINITY;
#pragma unroll
      for (int nf = 0; nf < 4; ++nf) {
        float s = sqk * ((float)cf[nf][r] - z_q * rk[nf] - z_k * rq[r] + zz);
        sv[nf] = s; smax = fmaxf(smax, s);
      }
      smax = fmaxf(smax, __shfl_xor(smax, 1, 32));
      smax = fmaxf(smax, __shfl_xor(smax, 2, 32));
      smax = fmaxf(smax, __shfl_xor(smax, 4, 32));
      smax = fmaxf(smax, __shfl_xor(smax, 8, 32));
      float mnew = fmaxf(mrow[r], smax);
      float ps = 0.f;
#pragma unroll
      for (int nf = 0; nf < 4; ++nf) ps += __expf(sv[nf] - mnew);
      ps += __shfl_xor(ps, 1, 32); ps += __shfl_xor(ps, 2, 32);
      ps += __shfl_xor(ps, 4, 32); ps += __shfl_xor(ps, 8, 32);
      lrow[r] = lrow[r] * __expf(mrow[r] - mnew) + ps;
      mrow[r] = mnew;
    }
  }

  v8i cpv[4] = {};
  int rsP[8] = {0, 0, 0, 0, 0, 0, 0, 0};
  uint8_t* sPw = sP + w * 16 * 64;
  const float iat = 1.f / a_at;
  for (int kt = 0; kt < NSEQ / 64; ++kt) {
    const int kb = kt * 64;
    __syncthreads();
    *(int4*)(sK + srow * 64 + soff) =
        *(const int4*)(kbase + (size_t)(kb + srow) * HD + soff);
    *(int4*)(sV + srow * 64 + soff) =
        *(const int4*)(vbase + (size_t)srow * NSEQ + kb + soff);
    __syncthreads();
    v8i cf[4];
#pragma unroll
    for (int nf = 0; nf < 4; ++nf) {
      v8i bk = load_frag_b(sK + nf * 16 * 64, 64, l16, half);
      v8i z = {};
      cf[nf] = __builtin_amdgcn_wmma_i32_16x16x64_iu8(true, aq, true, bk, z,
                                                      false, false);
    }
    float rk[4];
#pragma unroll
    for (int nf = 0; nf < 4; ++nf)
      rk[nf] = rsk[(size_t)bh * NSEQ + kb + nf * 16 + l16];
#pragma unroll
    for (int r = 0; r < 8; ++r) {
      int prs = 0;
#pragma unroll
      for (int nf = 0; nf < 4; ++nf) {
        float s = sqk * ((float)cf[nf][r] - z_q * rk[nf] - z_k * rq[r] + zz);
        float p = __expf(s - mrow[r]) / lrow[r];
        int pq = clampi((int)rintf(p * iat + z_at), 0, 15);
        sPw[(half * 8 + r) * 64 + nf * 16 + l16] = (uint8_t)pq;
        prs += pq;
      }
      prs += __shfl_xor(prs, 1, 32); prs += __shfl_xor(prs, 2, 32);
      prs += __shfl_xor(prs, 4, 32); prs += __shfl_xor(prs, 8, 32);
      rsP[r] += prs;
    }
    v8i ap = load_frag_a((const int8_t*)sPw, 64, l16, half);
#pragma unroll
    for (int nf = 0; nf < 4; ++nf) {
      v8i bv = load_frag_b(sV + nf * 16 * 64, 64, l16, half);
      cpv[nf] = __builtin_amdgcn_wmma_i32_16x16x64_iu8(false, ap, true, bv,
                                                       cpv[nf], false, false);
    }
  }

  const float av = a_at * a_v;
  const float zzp = (float)NSEQ * z_at * z_v;
  const float ip = 1.f / a_p;
#pragma unroll
  for (int nf = 0; nf < 4; ++nf) {
    int d = nf * 16 + l16;
    float cv = (float)csv[bh * HD + d];
#pragma unroll
    for (int r = 0; r < 8; ++r) {
      float x1 = av * ((float)cpv[nf][r] - z_at * cv - z_v * (float)rsP[r] + zzp);
      int xq = clampi((int)rintf(x1 * ip + z_p), -8, 7);
      int n = qrow0 + half * 8 + r;
      x1q[((size_t)b * NSEQ + n) * DIMC + h * HD + d] = (int8_t)xq;
    }
  }
}

extern "C" void kernel_launch(void* const* d_in, const int* in_sizes, int n_in,
                              void* d_out, int out_size, void* d_ws, size_t ws_size,
                              hipStream_t stream) {
  (void)in_sizes; (void)n_in; (void)out_size; (void)ws_size;
  const float* x0            = (const float*)d_in[0];
  const float* qkv_w         = (const float*)d_in[1];
  const float* qkv_alpha_w   = (const float*)d_in[2];
  const float* qkv_act_alpha = (const float*)d_in[3];
  const float* qkv_act_zp    = (const float*)d_in[4];
  const float* norm_q_g      = (const float*)d_in[5];
  const float* norm_q_b      = (const float*)d_in[6];
  const float* norm_k_g      = (const float*)d_in[7];
  const float* norm_k_b      = (const float*)d_in[8];
  const float* q_alpha       = (const float*)d_in[9];
  const float* q_zp          = (const float*)d_in[10];
  const float* k_alpha       = (const float*)d_in[11];
  const float* k_zp          = (const float*)d_in[12];
  const float* v_alpha       = (const float*)d_in[13];
  const float* v_zp          = (const float*)d_in[14];
  const float* attn_alpha    = (const float*)d_in[15];
  const float* attn_zp       = (const float*)d_in[16];
  const float* proj_w        = (const float*)d_in[17];
  const float* proj_b        = (const float*)d_in[18];
  const float* proj_alpha_w  = (const float*)d_in[19];
  const float* proj_act_alpha= (const float*)d_in[20];
  const float* proj_act_zp   = (const float*)d_in[21];
  float* out = (float*)d_out;

  char* ws = (char*)d_ws;
  size_t o = 0;
  auto take = [&](size_t bytes) -> char* {
    char* p = ws + o;
    o = (o + bytes + 255) & ~(size_t)255;
    return p;
  };
  float*  params  = (float*)take(64);
  int8_t* wq_qkv  = (int8_t*)take((size_t)3 * DIMC * DIMC);
  float*  cs_qkv  = (float*)take((size_t)3 * DIMC * 4);
  int8_t* wq_proj = (int8_t*)take((size_t)DIMC * DIMC);
  float*  cs_proj = (float*)take((size_t)DIMC * 4);
  int8_t* x0q     = (int8_t*)take((size_t)MROWS * DIMC);
  float*  qkvf    = (float*)take((size_t)MROWS * 3 * DIMC * 4);
  int8_t* q2i     = (int8_t*)take((size_t)BH * NSEQ * HD);
  int8_t* k2i     = (int8_t*)take((size_t)BH * NSEQ * HD);
  int8_t* v2t     = (int8_t*)take((size_t)BH * NSEQ * HD);
  float*  rsqv    = (float*)take((size_t)BH * NSEQ * 4);
  float*  rskv    = (float*)take((size_t)BH * NSEQ * 4);
  int*    csv     = (int*)take((size_t)BH * HD * 4);
  int8_t* x1q     = (int8_t*)take((size_t)MROWS * DIMC);

  scalars_kernel<<<1, 32, 0, stream>>>(qkv_act_alpha, qkv_act_zp, q_alpha, q_zp,
                                       k_alpha, k_zp, v_alpha, v_zp, attn_alpha,
                                       attn_zp, proj_act_alpha, proj_act_zp,
                                       params);
  quant_w<<<(3 * DIMC + DIMC) / 8, 256, 0, stream>>>(
      qkv_w, qkv_alpha_w, proj_w, proj_alpha_w, wq_qkv, cs_qkv, wq_proj, cs_proj);
  quant_x0<<<(MROWS * DIMC / 4 + 255) / 256, 256, 0, stream>>>(x0, params, 0,
                                                               x0q,
                                                               MROWS * DIMC / 4);
  gemm_q4<<<(MROWS / 128) * (3 * DIMC / 128), 256, 0, stream>>>(
      x0q, wq_qkv, qkv_alpha_w, cs_qkv, nullptr, qkvf, 3 * DIMC, DIMC, params, 0);
  ln_quant<<<BH * NSEQ / 8, 256, 0, stream>>>(qkvf, norm_q_g, norm_q_b, norm_k_g,
                                              norm_k_b, params, q2i, k2i, v2t,
                                              rsqv, rskv);
  colsum_v<<<BH * HD / 8, 256, 0, stream>>>(v2t, csv);
  attn_q4<<<BH * (NSEQ / 128), 256, 0, stream>>>(q2i, k2i, v2t, rsqv, rskv, csv,
                                                 params, x1q);
  gemm_q4<<<(MROWS / 128) * (DIMC / 128), 256, 0, stream>>>(
      x1q, wq_proj, proj_alpha_w, cs_proj, proj_b, out, DIMC, DIMC, params, 10);
}
